// EquivariantMessagePasser_47399259079026
// MI455X (gfx1250) — matrix operand order repaired and checked by reading, not compiled, not run on hardware
//
#include <hip/hip_runtime.h>

// ---------------------------------------------------------------------------
// CDNA5 / gfx1250 implementation of EquivariantMessagePasser.
// Heavy compute (per-pair radial MLP, ~1e11 FLOP) runs on v_wmma_f32_16x16x32_bf16.
// Block = 8 waves / 64 pairs: each wave owns a j-column tile and accumulates
// 4 pair sub-tiles in registers, so every streamed B-fragment feeds 4 WMMAs
// (4x less L2 weight traffic -- the previous binding constraint).
// Layer-3 of the MLP is fused with the gather/scatter message pass.
// ---------------------------------------------------------------------------

typedef __attribute__((ext_vector_type(16))) __bf16 v16bf;
typedef __attribute__((ext_vector_type(8)))  float  v8f;

union Frag {
    v16bf  v;
    float4 q[2];
};

__device__ __forceinline__ unsigned short f2bf(float x) {
    unsigned int u = __float_as_uint(x);
    u += 0x7FFFu + ((u >> 16) & 1u);   // round-to-nearest-even
    return (unsigned short)(u >> 16);
}

// Fast silu: v_exp + v_rcp + v_mul (avoid IEEE divide expansion).
__device__ __forceinline__ float silu(float x) {
    return x * __builtin_amdgcn_rcpf(1.0f + __expf(-x));
}

// A-operand fragment (16xK tile, activations) from an LDS bf16 matrix.
// Layout (ISA 7.12.2, 16-bit A 16x32): lanes 0-15 row=lane hold K=c0+0..7 and
// c0+16..23; lanes 16-31 row=lane-16 hold K=c0+8..15 and c0+24..31.
__device__ __forceinline__ v16bf load_afrag(const unsigned short* lds,
                                            int rowStride, int c0, int lane) {
    const int row = lane & 15;
    const int hi  = lane >> 4;
    const unsigned short* p = lds + row * rowStride + c0 + hi * 8;
    Frag f;
    f.q[0] = *(const float4*)(p);        // K = base + 0..7
    f.q[1] = *(const float4*)(p + 16);   // K = base + 16..23
    return f.v;
}

// B-operand fragment (32x16 tile of weights) from global bf16, stored
// TRANSPOSED: Wt[outCol][Kdim]. Lanes 0-15 col=j0+lane hold K=c0+0..7 and
// c0+16..23; lanes 16-31 col=j0+lane-16 hold K=c0+8..15 and c0+24..31.
__device__ __forceinline__ v16bf load_bfrag(const unsigned short* g,
                                            int rowStride, int j0, int c0, int lane) {
    const int col = j0 + (lane & 15);
    const int hi  = lane >> 4;
    const unsigned short* p = g + (size_t)col * rowStride + c0 + hi * 8;
    Frag f;
    f.q[0] = *(const float4*)(p);        // K = base + 0..7
    f.q[1] = *(const float4*)(p + 16);   // K = base + 16..23
    return f.v;
}

// ---------------------------------------------------------------------------
// Weight conversion: fp32 -> bf16 with transpose so B-operand loads are
// contiguous.  At[h][32] (zero-padded K=N..31), Bt[h][h], Ct[K][h].
// ---------------------------------------------------------------------------
template <int N, int K>
__global__ void __launch_bounds__(256)
convert_weights_kernel(const float* __restrict__ A, const float* __restrict__ B,
                       const float* __restrict__ C,
                       unsigned short* __restrict__ At,
                       unsigned short* __restrict__ Bt,
                       unsigned short* __restrict__ Ct) {
    constexpr int H = 4 * K;
    const int total = H * 32 + H * H + K * H;
    for (int i = blockIdx.x * blockDim.x + threadIdx.x; i < total;
         i += gridDim.x * blockDim.x) {
        if (i < H * 32) {
            int j = i >> 5, n = i & 31;                 // A: [N][H] -> At[j][n]
            At[i] = f2bf(n < N ? A[n * H + j] : 0.0f);
        } else if (i < H * 32 + H * H) {
            int idx = i - H * 32;
            int j = idx / H, k = idx % H;               // B: [H][H] -> Bt[j][k]
            Bt[idx] = f2bf(B[k * H + j]);
        } else {
            int idx = i - H * 32 - H * H;
            int j = idx / H, k = idx % H;               // C: [H][K] -> Ct[j][k]
            Ct[idx] = f2bf(C[k * K + j]);
        }
    }
}

// ---------------------------------------------------------------------------
// Fused radial MLP (WMMA bf16) + message pass.  64 pairs / block, 8 waves.
// ---------------------------------------------------------------------------
template <int M, int N, int K>
__global__ void __launch_bounds__(256)
radial_msg_kernel(const float* __restrict__ rb, const float* __restrict__ sh,
                  const float* __restrict__ U,
                  const unsigned short* __restrict__ At,
                  const unsigned short* __restrict__ Bt,
                  const unsigned short* __restrict__ Ct,
                  const float* __restrict__ x,      // [NATOMS, M, K]
                  float*       __restrict__ pooled, // [NATOMS, M, K]
                  const int* __restrict__ centers,
                  const int* __restrict__ neighbors, int P) {
    constexpr int H   = 4 * K;
    constexpr int HP  = H + 8;   // LDS row pad: rotate banks, keep 16B align
    constexpr int RBS = 40;      // 32 + 8 pad
    constexpr int PT  = 64;      // pairs per block (4 sub-tiles of 16)

    __shared__ unsigned short rbT[PT * RBS];
    __shared__ unsigned short h1[PT * HP];
    __shared__ unsigned short h2[PT * HP];
    __shared__ float          usL[PT * 8];
    __shared__ int            nbI[PT];
    __shared__ int            ctI[PT];

    const int tid  = threadIdx.x;
    const int lane = tid & 31;
    const int wave = tid >> 5;
    const int p0   = blockIdx.x * PT;

    // ---- stage rb tile (zero-padded to 32 K-cols) --------------------------
    for (int s = tid; s < PT * 32; s += 256) {
        int row = s >> 5, c = s & 31;
        int p = p0 + row;
        float v = (c < N && p < P) ? rb[(size_t)p * N + c] : 0.0f;
        rbT[row * RBS + c] = f2bf(v);
    }
    // ---- per-pair us[u] = sum_m U[u,m] * sh[p,m] ---------------------------
    for (int s = tid; s < PT * 8; s += 256) {
        int row = s >> 3, u = s & 7;
        int p = p0 + row;
        float acc = 0.0f;
        if (u < M && p < P) {
            for (int mm = 0; mm < M; ++mm) acc += U[u * M + mm] * sh[(size_t)p * M + mm];
        }
        usL[s] = acc;
    }
    if (tid < PT) {
        int p = p0 + tid;
        nbI[tid] = (p < P) ? neighbors[p] : 0;
        ctI[tid] = (p < P) ? centers[p]   : 0;
    }
    __syncthreads();

    // ---- layer 1: [PTxN(pad32)] @ A[N x H] -> silu -> h1 -------------------
    {
        v16bf aF[4];
        #pragma unroll
        for (int t = 0; t < 4; ++t)
            aF[t] = load_afrag(rbT + t * 16 * RBS, RBS, 0, lane);
        for (int j0 = wave * 16; j0 < H; j0 += 128) {
            v16bf bF = load_bfrag(At, 32, j0, 0, lane);
            int col = j0 + (lane & 15);
            int rb0 = (lane >> 4) * 8;
            #pragma unroll
            for (int t = 0; t < 4; ++t) {
                v8f acc = {};
                acc = __builtin_amdgcn_wmma_f32_16x16x32_bf16(
                    false, aF[t], false, bF, (short)0, acc, false, false);
                #pragma unroll
                for (int i = 0; i < 8; ++i)
                    h1[(t * 16 + rb0 + i) * HP + col] = f2bf(silu(acc[i]));
            }
        }
    }
    __syncthreads();

    // ---- layer 2: h1 @ B[H x H] -> silu -> h2 ------------------------------
    for (int j0 = wave * 16; j0 < H; j0 += 128) {
        v8f acc[4] = {};
        for (int c0 = 0; c0 < H; c0 += 32) {
            v16bf bF = load_bfrag(Bt, H, j0, c0, lane);
            #pragma unroll
            for (int t = 0; t < 4; ++t) {
                v16bf aF = load_afrag(h1 + t * 16 * HP, HP, c0, lane);
                acc[t] = __builtin_amdgcn_wmma_f32_16x16x32_bf16(
                    false, aF, false, bF, (short)0, acc[t], false, false);
            }
        }
        int col = j0 + (lane & 15);
        int rb0 = (lane >> 4) * 8;
        #pragma unroll
        for (int t = 0; t < 4; ++t)
            #pragma unroll
            for (int i = 0; i < 8; ++i)
                h2[(t * 16 + rb0 + i) * HP + col] = f2bf(silu(acc[t][i]));
    }
    __syncthreads();

    // ---- layer 3: h2 @ C[H x K], fused with message scatter ----------------
    for (int j0 = wave * 16; j0 < K; j0 += 128) {
        v8f acc[4] = {};
        for (int c0 = 0; c0 < H; c0 += 32) {
            v16bf bF = load_bfrag(Ct, H, j0, c0, lane);
            #pragma unroll
            for (int t = 0; t < 4; ++t) {
                v16bf aF = load_afrag(h2 + t * 16 * HP, HP, c0, lane);
                acc[t] = __builtin_amdgcn_wmma_f32_16x16x32_bf16(
                    false, aF, false, bF, (short)0, acc[t], false, false);
            }
        }
        int col = j0 + (lane & 15);
        int rb0 = (lane >> 4) * 8;
        #pragma unroll
        for (int t = 0; t < 4; ++t) {
            #pragma unroll
            for (int i = 0; i < 8; ++i) {
                int row = t * 16 + rb0 + i;
                int p   = p0 + row;
                if (p < P) {
                    float rv = acc[t][i];
                    const float* xr = x + (size_t)nbI[row] * (M * K);
                    float* pr = pooled + (size_t)ctI[row] * (M * K);
                    #pragma unroll
                    for (int u = 0; u < M; ++u) {
                        float s = 0.1f * usL[row * 8 + u];
                        atomicAdd(&pr[u * K + col], s * rv * xr[u * K + col]);
                    }
                }
            }
        }
    }
}

// ---------------------------------------------------------------------------
// Per-atom transform: out = [resid +] U * ((U^T * maybe_rmsnorm(in)) @ W)
// Small (<=320 elems/atom); plain VALU.
// ---------------------------------------------------------------------------
template <int M, int K, bool RMS, bool RESID>
__global__ void __launch_bounds__(128)
node_linear_kernel(const float* __restrict__ fin,  // [NA, M, K] transform input
                   const float* __restrict__ fres, // residual (RESID)
                   const float* __restrict__ U,    // [M, M]
                   const float* __restrict__ g,    // [K] (RMS)
                   const float* __restrict__ W,    // [K, K]
                   float* __restrict__ out, int natoms) {
    constexpr int E = M * K;
    __shared__ float y[E];
    __shared__ float c[E];
    __shared__ float red[4];

    const int a = blockIdx.x;
    const int t = threadIdx.x;
    if (a >= natoms) return;
    const float* fa = fin + (size_t)a * E;

    if (RMS) {
        float ss = 0.0f;
        for (int e = t; e < E; e += 128) { float v = fa[e]; ss += v * v; }
        for (int off = 16; off > 0; off >>= 1) ss += __shfl_down(ss, off, 32);
        if ((t & 31) == 0) red[t >> 5] = ss;
        __syncthreads();
        float rms = rsqrtf((red[0] + red[1] + red[2] + red[3]) / (float)E + 1e-6f);
        for (int e = t; e < E; e += 128) y[e] = fa[e] * rms * g[e % K];
    } else {
        for (int e = t; e < E; e += 128) y[e] = fa[e];
    }
    __syncthreads();
    // c[m,k] = sum_u U[u,m] * y[u,k]   (couple, U^T)
    for (int e = t; e < E; e += 128) {
        int mm = e / K, k = e % K;
        float acc = 0.0f;
        for (int u = 0; u < M; ++u) acc += U[u * M + mm] * y[u * K + k];
        c[e] = acc;
    }
    __syncthreads();
    // c2[m,j] = sum_k c[m,k] * W[k,j]  (channel mix) -> reuse y
    for (int e = t; e < E; e += 128) {
        int mm = e / K, j = e % K;
        float acc = 0.0f;
        for (int k = 0; k < K; ++k) acc += c[mm * K + k] * W[k * K + j];
        y[e] = acc;
    }
    __syncthreads();
    // out[u,k] = resid + sum_m U[u,m] * c2[m,k]  (uncouple)
    for (int e = t; e < E; e += 128) {
        int u = e / K, k = e % K;
        float acc = 0.0f;
        for (int mm = 0; mm < M; ++mm) acc += U[u * M + mm] * y[mm * K + k];
        if (RESID) acc += fres[(size_t)a * E + e];
        out[(size_t)a * E + e] = acc;
    }
}

// ---------------------------------------------------------------------------
// Host launcher.
// Inputs (dict order): per l: rb, sh, f, U, g, Win, Wout, A, B, C ; then
// centers (d_in[40]), neighbors (d_in[41]).
// ws layout (floats): x (NA*960) | pooled (NA*960) | bf16 weights (~1.3MB)
// ---------------------------------------------------------------------------
extern "C" void kernel_launch(void* const* d_in, const int* in_sizes, int n_in,
                              void* d_out, int out_size, void* d_ws, size_t ws_size,
                              hipStream_t stream) {
    (void)n_in; (void)out_size; (void)ws_size;

    const int natoms = in_sizes[2] / 128;  // f0 = [NA, 1, 128]
    const int npairs = in_sizes[0] / 8;    // rb0 = [P, 8]

    const int* centers   = (const int*)d_in[40];
    const int* neighbors = (const int*)d_in[41];

    const size_t XTOT = (size_t)natoms * (1 * 128 + 3 * 96 + 5 * 64 + 7 * 32);
    float* xbase = (float*)d_ws;
    float* pbase = xbase + XTOT;
    unsigned short* wbase = (unsigned short*)(pbase + XTOT);

    // zero the pooled accumulator (atomics target)
    hipMemsetAsync(pbase, 0, XTOT * sizeof(float), stream);

    size_t xoff = 0, woff = 0, ooff = 0;

#define LAUNCH_LEVEL(l, Mv, Nv, Kv)                                                \
    {                                                                              \
        constexpr int Hc = 4 * (Kv);                                               \
        const float* rb   = (const float*)d_in[(l) * 10 + 0];                      \
        const float* sh   = (const float*)d_in[(l) * 10 + 1];                      \
        const float* f    = (const float*)d_in[(l) * 10 + 2];                      \
        const float* U    = (const float*)d_in[(l) * 10 + 3];                      \
        const float* g    = (const float*)d_in[(l) * 10 + 4];                      \
        const float* Win  = (const float*)d_in[(l) * 10 + 5];                      \
        const float* Wout = (const float*)d_in[(l) * 10 + 6];                      \
        const float* A    = (const float*)d_in[(l) * 10 + 7];                      \
        const float* B    = (const float*)d_in[(l) * 10 + 8];                      \
        const float* C    = (const float*)d_in[(l) * 10 + 9];                      \
        float* xL = xbase + xoff;                                                  \
        float* pL = pbase + xoff;                                                  \
        unsigned short* At = wbase + woff;                                         \
        unsigned short* Bt = At + (size_t)Hc * 32;                                 \
        unsigned short* Ct = Bt + (size_t)Hc * Hc;                                 \
        float* outL = (float*)d_out + ooff;                                        \
        convert_weights_kernel<Nv, Kv><<<256, 256, 0, stream>>>(A, B, C, At, Bt, Ct); \
        node_linear_kernel<Mv, Kv, true, false><<<natoms, 128, 0, stream>>>(       \
            f, nullptr, U, g, Win, xL, natoms);                                    \
        radial_msg_kernel<Mv, Nv, Kv><<<(npairs + 63) / 64, 256, 0, stream>>>(     \
            rb, sh, U, At, Bt, Ct, xL, pL, centers, neighbors, npairs);            \
        node_linear_kernel<Mv, Kv, false, true><<<natoms, 128, 0, stream>>>(       \
            pL, f, U, nullptr, Wout, outL, natoms);                                \
        xoff += (size_t)natoms * (Mv) * (Kv);                                      \
        woff += (size_t)Hc * 32 + (size_t)Hc * Hc + (size_t)(Kv) * Hc;             \
        ooff += (size_t)natoms * (Mv) * (Kv);                                      \
    }

    LAUNCH_LEVEL(0, 1, 8, 128)
    LAUNCH_LEVEL(1, 3, 8, 96)
    LAUNCH_LEVEL(2, 5, 6, 64)
    LAUNCH_LEVEL(3, 7, 4, 32)

#undef LAUNCH_LEVEL
}
